// GroupedQueryAttention_32169305047119
// MI455X (gfx1250) — matrix operand order repaired and checked by reading, not compiled, not run on hardware
//
#include <hip/hip_runtime.h>

// Problem constants: B=2, N=2048, E=2048, H=16, G=4, HK=4, D=128, WIN=256
static constexpr int   BN    = 4096;         // B*N rows
static constexpr int   LDQKV = 3072;         // H*D + 2*HK*D
static constexpr float SCALE = 0.08838834764831845f; // 1/sqrt(128)

typedef __attribute__((ext_vector_type(16))) __bf16 v16bf;
typedef __attribute__((ext_vector_type(8)))  float  v8f;

struct FragU { unsigned u[8]; };

__device__ __forceinline__ v16bf frag_cast(const FragU& f) {
  return __builtin_bit_cast(v16bf, f);
}
__device__ __forceinline__ unsigned short f2bf(float f) {
  unsigned u = __builtin_bit_cast(unsigned, f);
  u += 0x7fffu + ((u >> 16) & 1u);           // round-to-nearest-even
  return (unsigned short)(u >> 16);
}
__device__ __forceinline__ float bf2f(unsigned short h) {
  unsigned u = ((unsigned)h) << 16;
  return __builtin_bit_cast(float, u);
}
__device__ __forceinline__ unsigned pk2(float lo, float hi) {
  return (unsigned)f2bf(lo) | ((unsigned)f2bf(hi) << 16);
}
__device__ __forceinline__ v8f wmma_bf16(v16bf a, v16bf b, v8f c) {
  return __builtin_amdgcn_wmma_f32_16x16x32_bf16(false, a, false, b, (short)0, c,
                                                 false, false);
}
// A-matrix (16-bit, 16x32) K index held by VGPR j for lane-half `half`
__device__ __forceinline__ int a_kmap(int j, int half) {
  return half * 16 + 2 * j;
}
__device__ __forceinline__ v8f vzero() {
  v8f v;
#pragma unroll
  for (int r = 0; r < 8; ++r) v[r] = 0.0f;
  return v;
}

// ---- CDNA5 async global->LDS DMA (ASYNCcnt-tracked), GV addressing mode ----
__device__ __forceinline__ void async_b128(unsigned lds_byte, const void* g) {
  asm volatile("global_load_async_to_lds_b128 %0, %1, off"
               :: "v"(lds_byte), "v"((unsigned long long)(size_t)g)
               : "memory");
}
__device__ __forceinline__ void wait_async0() {
#if __has_builtin(__builtin_amdgcn_s_wait_asynccnt)
  __builtin_amdgcn_s_wait_asynccnt(0);
#else
  asm volatile("s_wait_asynccnt 0" ::: "memory");
#endif
}
__device__ __forceinline__ unsigned lds_off(const void* p) {
  return (unsigned)(size_t)p;   // low 32 bits of generic ptr = group-segment offset
}

// ---------------------------------------------------------------------------
// Elementwise f32 -> bf16 (row-major copy, used for x)
// ---------------------------------------------------------------------------
__global__ __launch_bounds__(256) void cvt_f32_bf16(const float* __restrict__ src,
                                                    unsigned short* __restrict__ dst,
                                                    long long n) {
  long long i = (long long)blockIdx.x * 256 + threadIdx.x;
  if (i < n) dst[i] = f2bf(src[i]);
}

// Wq|Wk|Wv -> TRANSPOSED bf16 Wt[3072 n][2048 k] (coalesced writes)
__global__ __launch_bounds__(256) void cvt_wqkv_t(const float* __restrict__ Wq,
                                                  const float* __restrict__ Wk,
                                                  const float* __restrict__ Wv,
                                                  unsigned short* __restrict__ dst) {
  long long i = (long long)blockIdx.x * 256 + threadIdx.x;
  if (i >= (long long)3072 * 2048) return;
  const int k = (int)(i & 2047);
  const int n = (int)(i >> 11);
  float v;
  if (n < 2048)      v = Wq[(size_t)k * 2048 + n];
  else if (n < 2560) v = Wk[(size_t)k * 512 + (n - 2048)];
  else               v = Wv[(size_t)k * 512 + (n - 2560)];
  dst[i] = f2bf(v);
}

// Wo -> TRANSPOSED bf16 Wt[2048 n][2048 k]
__global__ __launch_bounds__(256) void cvt_wo_t(const float* __restrict__ Wo,
                                                unsigned short* __restrict__ dst) {
  long long i = (long long)blockIdx.x * 256 + threadIdx.x;
  if (i >= (long long)2048 * 2048) return;
  const int k = (int)(i & 2047);
  const int n = (int)(i >> 11);
  dst[i] = f2bf(Wo[(size_t)k * 2048 + n]);
}

// ---------------------------------------------------------------------------
// WMMA bf16 GEMM: C[M,N] = A[M,K] * Bt[N,K]^T. 128x128 block, 8 waves (2x4),
// wave = 64x32 via 4x2 accum tiles. BLOCK_K = 64. Async-DMA double-buffered
// LDS; both operands row-major-in-k so all fragment reads are b32-contiguous.
// ---------------------------------------------------------------------------
template <bool OUT_BF16>
__global__ __launch_bounds__(256) void gemm_bf16(const unsigned short* __restrict__ A,
                                                 const unsigned short* __restrict__ Bt,
                                                 void* __restrict__ Cv,
                                                 int M, int Nn, int K) {
  constexpr int LDT = 72;   // padded row stride: 144B = 16B-aligned, bank step 36
  __shared__ __align__(16) unsigned short As[2][128 * LDT];
  __shared__ __align__(16) unsigned short Bs[2][128 * LDT];
  const int tid  = threadIdx.x;
  const int lane = tid & 31, wid = tid >> 5;
  const int half = lane >> 4, ln = lane & 15;
  const int bm = blockIdx.y * 128, bn = blockIdx.x * 128;
  const int wm = (wid & 1) * 64, wn = (wid >> 1) * 32;

  v8f acc[4][2];
#pragma unroll
  for (int mi = 0; mi < 4; ++mi)
#pragma unroll
    for (int ni = 0; ni < 2; ++ni) acc[mi][ni] = vzero();

  const int crow = tid >> 1, ccb = (tid & 1) * 32;   // 32 elems (64B) per thread

  auto issue_tile = [&](int buf, int k0) {
    const unsigned short* ga = A + (size_t)(bm + crow) * K + k0 + ccb;
    unsigned la = lds_off(&As[buf][crow * LDT + ccb]);
#pragma unroll
    for (int i = 0; i < 4; ++i) async_b128(la + 16 * i, ga + 8 * i);
    const unsigned short* gb = Bt + (size_t)(bn + crow) * K + k0 + ccb;
    unsigned lb = lds_off(&Bs[buf][crow * LDT + ccb]);
#pragma unroll
    for (int i = 0; i < 4; ++i) async_b128(lb + 16 * i, gb + 8 * i);
  };

  issue_tile(0, 0);
  const int nk = K >> 6;
  for (int ki = 0; ki < nk; ++ki) {
    const int cur = ki & 1;
    wait_async0();
    __syncthreads();
    if (ki + 1 < nk) issue_tile(cur ^ 1, (ki + 1) * 64);
    if (ki + 2 < nk)
      __builtin_prefetch(A + (size_t)(bm + crow) * K + (ki + 2) * 64 + ccb, 0, 3);

#pragma unroll
    for (int c = 0; c < 2; ++c) {           // two K=32 steps per LDS tile
      FragU bfu;
      v16bf bfr[2];
#pragma unroll
      for (int ni = 0; ni < 2; ++ni) {
        const int n = wn + ni * 16 + ln;
#pragma unroll
        for (int j = 0; j < 8; ++j)
          bfu.u[j] = *(const unsigned*)&Bs[cur][n * LDT + c * 32 + half * 16 + 2 * j];
        bfr[ni] = frag_cast(bfu);
      }
#pragma unroll
      for (int mi = 0; mi < 4; ++mi) {
        FragU af;
        const int m = wm + mi * 16 + ln;
#pragma unroll
        for (int j = 0; j < 8; ++j)
          af.u[j] = *(const unsigned*)&As[cur][m * LDT + c * 32 + a_kmap(j, half)];
        v16bf afr = frag_cast(af);
        acc[mi][0] = wmma_bf16(afr, bfr[0], acc[mi][0]);
        acc[mi][1] = wmma_bf16(afr, bfr[1], acc[mi][1]);
      }
    }
  }

#pragma unroll
  for (int mi = 0; mi < 4; ++mi)
#pragma unroll
    for (int ni = 0; ni < 2; ++ni) {
      const int col = bn + wn + ni * 16 + ln;
#pragma unroll
      for (int r = 0; r < 8; ++r) {
        const int row = bm + wm + mi * 16 + half * 8 + r;
        const float v = acc[mi][ni][r];
        if (OUT_BF16)
          ((unsigned short*)Cv)[(size_t)row * Nn + col] = f2bf(v);
        else
          ((float*)Cv)[(size_t)row * Nn + col] = v;
      }
    }
}

// ---------------------------------------------------------------------------
// In-place RoPE on bf16 qkv buffer (q cols [0,2048), k cols [2048,2560))
// ---------------------------------------------------------------------------
__global__ __launch_bounds__(256) void rope_kernel(unsigned short* __restrict__ qkv,
                                                   const float* __restrict__ cosb,
                                                   const float* __restrict__ sinb) {
  long long idx = (long long)blockIdx.x * 256 + threadIdx.x;
  const long long total = (long long)BN * 20 * 64;
  if (idx >= total) return;
  const int  m    = (int)(idx % 64);
  long long  t2   = idx / 64;
  const int  hall = (int)(t2 % 20);
  const int  tok  = (int)(t2 / 20);
  const int  col  = (hall < 16) ? (hall * 128 + 2 * m)
                                : (2048 + (hall - 16) * 128 + 2 * m);
  unsigned short* p = qkv + (long long)tok * LDQKV + col;
  const float x0 = bf2f(p[0]), x1 = bf2f(p[1]);
  const float c = cosb[(long long)tok * 64 + m];
  const float s = sinb[(long long)tok * 64 + m];
  p[0] = f2bf(x0 * c - x1 * s);
  p[1] = f2bf(x0 * s + x1 * c);
}

// Extract V transposed: vT[((b*HK+hk)*128 + d)*2048 + t] = V[b,t,hk,d]
__global__ __launch_bounds__(256) void extract_vT(const unsigned short* __restrict__ qkv,
                                                  unsigned short* __restrict__ vT) {
  long long i = (long long)blockIdx.x * 256 + threadIdx.x;
  if (i >= (long long)8 * 128 * 2048) return;
  const int t  = (int)(i & 2047);
  const int d  = (int)((i >> 11) & 127);
  const int bh = (int)(i >> 18);
  const int b = bh >> 2, hk = bh & 3;
  vT[i] = qkv[((size_t)b * 2048 + t) * LDQKV + 2560 + hk * 128 + d];
}

// ---------------------------------------------------------------------------
// Windowed-causal flash attention. Grid: (N/16, B*HK), block = 128 (4 waves).
// Async-DMA double-buffered K (row-major) and Vᵀ (d-major) tiles; all
// fragment reads are contiguous b32. Sᵀ = K·Qᵀ, Oᵀ = Vᵀ·Pᵀ.
// ---------------------------------------------------------------------------
__global__ __launch_bounds__(128) void attn_kernel(const unsigned short* __restrict__ qkv,
                                                   const unsigned short* __restrict__ vT,
                                                   unsigned short* __restrict__ attn_out) {
  constexpr int LDK = 136;  // K tile rows: 272B, 16B-aligned, bank step 4
  constexpr int LDV = 40;   // Vᵀ tile rows: 80B, 16B-aligned, bank step 20
  __shared__ __align__(16) unsigned short Ks[2][32 * LDK];    // [key][d]
  __shared__ __align__(16) unsigned short VTs[2][128 * LDV];  // [d][key]
  const int tid  = threadIdx.x;
  const int lane = tid & 31, g = tid >> 5;
  const int half = lane >> 4, ln = lane & 15;
  const int q0 = blockIdx.x * 16;
  const int bh = blockIdx.y;                // b*HK + hk
  const int b  = bh >> 2, hk = bh & 3;
  const int h  = hk * 4 + g;
  const size_t tok0 = (size_t)b * 2048;
  const int iq = q0 + ln;

  // Preload Q B-fragments (Qᵀ: 32 d-rows x 16 q-cols), 4 d-chunks of 32
  v16bf bq[4];
  {
    const unsigned short* qrow = qkv + (tok0 + q0 + ln) * LDQKV + h * 128;
#pragma unroll
    for (int c = 0; c < 4; ++c) {
      FragU f;
#pragma unroll
      for (int j = 0; j < 8; ++j)
        f.u[j] = *(const unsigned*)&qrow[c * 32 + half * 16 + 2 * j];
      bq[c] = frag_cast(f);
    }
  }

  v8f o[8];
#pragma unroll
  for (int t = 0; t < 8; ++t) o[t] = vzero();
  float m_run = -1e30f, l_run = 0.0f;

  int lo = q0 - 255;
  if (lo < 0) lo = 0;
  const int kt0 = lo & ~31;
  const int kt1 = (q0 + 15) & ~31;
  const int nt  = ((kt1 - kt0) >> 5) + 1;

  const int ckey = tid >> 2, ccb = (tid & 3) * 32;  // K tile: 32 elems/thread
  const unsigned short* vbase = vT + (size_t)bh * 128 * 2048;

  auto issue_kv = [&](int buf, int kt) {
    const unsigned short* gk =
        qkv + (tok0 + kt + ckey) * LDQKV + 2048 + hk * 128 + ccb;
    unsigned lk = lds_off(&Ks[buf][ckey * LDK + ccb]);
#pragma unroll
    for (int i = 0; i < 4; ++i) async_b128(lk + 16 * i, gk + 8 * i);
    // Vᵀ tile: row d = tid, 32 keys (64B)
    const unsigned short* gv = vbase + (size_t)tid * 2048 + kt;
    unsigned lv = lds_off(&VTs[buf][tid * LDV]);
#pragma unroll
    for (int i = 0; i < 4; ++i) async_b128(lv + 16 * i, gv + 8 * i);
  };

  issue_kv(0, kt0);
  for (int it = 0; it < nt; ++it) {
    const int kt = kt0 + it * 32, cur = it & 1;
    wait_async0();
    __syncthreads();
    if (it + 1 < nt) issue_kv(cur ^ 1, kt + 32);

    // Sᵀ = K·Qᵀ for two 16-key subtiles, accumulated over 4 d-chunks
    v8f s[2];
#pragma unroll
    for (int si = 0; si < 2; ++si) {
      s[si] = vzero();
      const int key = si * 16 + ln;
#pragma unroll
      for (int c = 0; c < 4; ++c) {
        FragU kf;
#pragma unroll
        for (int j = 0; j < 8; ++j)
          kf.u[j] = *(const unsigned*)&Ks[cur][key * LDK + c * 32 + a_kmap(j, half)];
        s[si] = wmma_bf16(frag_cast(kf), bq[c], s[si]);
      }
    }

    // Scale + sliding-window causal mask; online softmax update
    float p[16];
    float tmax = -1e30f;
#pragma unroll
    for (int si = 0; si < 2; ++si)
#pragma unroll
      for (int r = 0; r < 8; ++r) {
        const int key = kt + si * 16 + half * 8 + r;
        float v = s[si][r] * SCALE;
        const bool ok = (key <= iq) && (key > iq - 256);
        v = ok ? v : -1e30f;
        p[si * 8 + r] = v;
        tmax = fmaxf(tmax, v);
      }
    tmax = fmaxf(tmax, __shfl_xor(tmax, 16, 32));
    const float mnew = fmaxf(m_run, tmax);
    const float corr = __expf(m_run - mnew);
    l_run *= corr;
#pragma unroll
    for (int t = 0; t < 8; ++t)
#pragma unroll
      for (int r = 0; r < 8; ++r) o[t][r] *= corr;
    float psum = 0.0f;
#pragma unroll
    for (int i = 0; i < 16; ++i) {
      p[i] = __expf(p[i] - mnew);
      psum += p[i];
    }
    psum += __shfl_xor(psum, 16, 32);
    l_run += psum;
    m_run = mnew;

    // Repack Pᵀ (C-layout) into B-layout over 32 keys via half-exchange
    float x0[8], x1[8];
#pragma unroll
    for (int r = 0; r < 8; ++r) {
      x0[r] = __shfl_xor(p[r], 16, 32);
      x1[r] = __shfl_xor(p[8 + r], 16, 32);
    }
    FragU pf;
#pragma unroll
    for (int j = 0; j < 4; ++j)
      pf.u[j] = half ? pk2(x1[2 * j], x1[2 * j + 1])
                     : pk2(p[2 * j], p[2 * j + 1]);
#pragma unroll
    for (int j = 0; j < 4; ++j)
      pf.u[4 + j] = half ? pk2(p[8 + 2 * j], p[8 + 2 * j + 1])
                         : pk2(x0[2 * j], x0[2 * j + 1]);
    const v16bf bp = frag_cast(pf);

    // Oᵀ += Vᵀ·Pᵀ : contiguous pair reads from d-major Vᵀ tile
#pragma unroll
    for (int t = 0; t < 8; ++t) {
      FragU vf;
      const int dr = t * 16 + ln;
#pragma unroll
      for (int j = 0; j < 8; ++j)
        vf.u[j] = *(const unsigned*)&VTs[cur][dr * LDV + a_kmap(j, half)];
      o[t] = wmma_bf16(frag_cast(vf), bp, o[t]);
    }
  }

  const float inv = 1.0f / l_run;
  unsigned short* orow = attn_out + (tok0 + q0 + ln) * 2048 + h * 128;
#pragma unroll
  for (int t = 0; t < 8; ++t)
#pragma unroll
    for (int r = 0; r < 8; ++r)
      orow[t * 16 + half * 8 + r] = f2bf(o[t][r] * inv);
}

// ---------------------------------------------------------------------------
extern "C" void kernel_launch(void* const* d_in, const int* in_sizes, int n_in,
                              void* d_out, int out_size, void* d_ws, size_t ws_size,
                              hipStream_t stream) {
  (void)in_sizes; (void)n_in; (void)out_size; (void)ws_size;
  const float* x  = (const float*)d_in[0];
  const float* rc = (const float*)d_in[1];
  const float* rs = (const float*)d_in[2];
  const float* Wq = (const float*)d_in[3];
  const float* Wk = (const float*)d_in[4];
  const float* Wv = (const float*)d_in[5];
  const float* Wo = (const float*)d_in[6];
  float* out = (float*)d_out;

  char* ws = (char*)d_ws;
  unsigned short* xb    = (unsigned short*)(ws + 0);          // 16,777,216 B
  unsigned short* wqkvT = (unsigned short*)(ws + 16777216);   // 12,582,912 B
  unsigned short* woT   = (unsigned short*)(ws + 29360128);   //  8,388,608 B
  unsigned short* qkv   = (unsigned short*)(ws + 37748736);   // 25,165,824 B
  unsigned short* attn  = (unsigned short*)(ws + 62914560);   // 16,777,216 B
  unsigned short* vT    = (unsigned short*)(ws + 79691776);   //  4,194,304 B

  cvt_f32_bf16<<<(8388608 + 255) / 256, 256, 0, stream>>>(x, xb, 8388608LL);
  cvt_wqkv_t<<<(6291456 + 255) / 256, 256, 0, stream>>>(Wq, Wk, Wv, wqkvT);
  cvt_wo_t<<<(4194304 + 255) / 256, 256, 0, stream>>>(Wo, woT);

  gemm_bf16<true><<<dim3(3072 / 128, 4096 / 128), 256, 0, stream>>>(
      xb, wqkvT, (void*)qkv, BN, LDQKV, 2048);

  rope_kernel<<<(5242880 + 255) / 256, 256, 0, stream>>>(qkv, rc, rs);
  extract_vT<<<(2097152 + 255) / 256, 256, 0, stream>>>(qkv, vT);

  attn_kernel<<<dim3(2048 / 16, 2 * 4), 128, 0, stream>>>(qkv, vT, attn);

  gemm_bf16<false><<<dim3(2048 / 128, 4096 / 128), 256, 0, stream>>>(
      attn, woT, (void*)out, BN, 2048, 2048);
}
